// VisionMamba_21431886807225
// MI455X (gfx1250) — matrix operand order repaired and checked by reading, not compiled, not run on hardware
//
#include <hip/hip_runtime.h>
#include <hip/hip_bf16.h>
#include <math.h>

// ---------------- model constants ----------------
#define BATCH_   8
#define SEQ      1568          // 14*14*8 tokens per image
#define NTOK     12544         // BATCH_*SEQ
#define DM       768
#define DI       1536
#define DSTATE   16
#define DTRANK   48
#define DCONV    4
#define NCLS     1000
#define LAYERS   4
#define EPS_     1e-5f

typedef __bf16 bf16_t;
typedef __attribute__((ext_vector_type(16))) __bf16 v16bf;
typedef __attribute__((ext_vector_type(8)))  __bf16 v8bf;
typedef __attribute__((ext_vector_type(8)))  float  v8f;

__device__ __forceinline__ bf16_t f2bf(float f) { return (bf16_t)f; }
__device__ __forceinline__ float  siluf(float v) { return v / (1.0f + expf(-v)); }

// ---------------------------------------------------------------------------
// WMMA bf16 GEMM:  C[M,N] = A[M,K] * W[N,K]^T (+bias)
// A row-major bf16, W row-major bf16 (row = output col), C row-major f32.
// Requires M%16==0, K%32==0 (K%128==0 for the unroll), rows 32-byte aligned.
// Block = 256 threads (8 waves); wave w -> row tile blockIdx.y*8+w,
// each wave produces a 16 x (NT*16) strip; NT is compile-time so the K-loop
// is branch-free and accumulators stay pinned in registers.
// ---------------------------------------------------------------------------
template<int NT>
__global__ __launch_bounds__(256)
void vm_gemm_bf16_wmma(const bf16_t* __restrict__ A, const bf16_t* __restrict__ W,
                       const float* __restrict__ bias, float* __restrict__ C,
                       int M, int N, int K, int n_base)
{
    const int lane = threadIdx.x & 31;
    const int wave = threadIdx.x >> 5;
    const int mt   = blockIdx.y * 8 + wave;
    if (mt * 16 >= M) return;                 // wave-uniform exit
    const int hi = lane >> 4;                 // lane group (0/1)
    const int lm = lane & 15;
    const int n0 = n_base + blockIdx.x * (NT * 16);

    const bf16_t* Arow = A + (size_t)(mt * 16 + lm) * K;
    const bf16_t* Wp[NT];
#pragma unroll
    for (int t = 0; t < NT; ++t)
        Wp[t] = W + (size_t)(n0 + t * 16 + lm) * K;

    v8f acc[NT];
#pragma unroll
    for (int t = 0; t < NT; ++t) acc[t] = v8f{};

#pragma unroll 4
    for (int k0 = 0; k0 < K; k0 += 32) {
        // A fragment: e0..15 -> k = k0+hi*16+e
        union { v16bf v; v8bf h[2]; } a;
        a.h[0] = *(const v8bf*)(Arow + k0 + hi * 16);
        a.h[1] = *(const v8bf*)(Arow + k0 + hi * 16 + 8);
#pragma unroll
        for (int t = 0; t < NT; ++t) {
            // B fragment: lane holds col n, 16 contiguous k starting at k0+hi*16
            v16bf b = *(const v16bf*)(Wp[t] + k0 + hi * 16);
            acc[t] = __builtin_amdgcn_wmma_f32_16x16x32_bf16(
                false, a.v, false, b, (short)0, acc[t], false, false);
        }
    }

#pragma unroll
    for (int t = 0; t < NT; ++t) {
        const int col = n0 + t * 16 + lm;
        const float bv = bias ? bias[col] : 0.0f;
#pragma unroll
        for (int v = 0; v < 8; ++v) {
            const int row = mt * 16 + hi * 8 + v;   // D layout: VGPR v -> M=v(+8*hi)
            C[(size_t)row * N + col] = acc[t][v] + bv;
        }
    }
}

// ---------------------------------------------------------------------------
// elementwise / small kernels
// ---------------------------------------------------------------------------
__global__ void vm_f32_to_bf16(const float* __restrict__ in, bf16_t* __restrict__ out, size_t n)
{
    size_t i = (size_t)blockIdx.x * blockDim.x + threadIdx.x;
    if (i < n) out[i] = f2bf(in[i]);
}

__global__ void vm_add_inplace(float* __restrict__ a, const float* __restrict__ b, size_t n)
{
    size_t i = (size_t)blockIdx.x * blockDim.x + threadIdx.x;
    if (i < n) a[i] += b[i];
}

// im2col: A[r, p*16+q] = x[b, c, h*16+p, w*16+q], token r = b*SEQ + ((h*14+w)*8+c)
__global__ void vm_im2col(const float* __restrict__ x, bf16_t* __restrict__ Ap)
{
    size_t idx = (size_t)blockIdx.x * blockDim.x + threadIdx.x;
    if (idx >= (size_t)NTOK * 256) return;
    int pq = idx & 255;
    int r  = idx >> 8;
    int p = pq >> 4, q = pq & 15;
    int b = r / SEQ;
    int t = r % SEQ;
    int c  = t & 7;
    int hw = t >> 3;
    int h = hw / 14, w = hw % 14;
    size_t src = (((size_t)(b * 8 + c) * 224) + (size_t)(h * 16 + p)) * 224 + (w * 16 + q);
    Ap[idx] = f2bf(x[src]);
}

__global__ void vm_add_channel_embed(float* __restrict__ resid, const float* __restrict__ ce)
{
    size_t idx = (size_t)blockIdx.x * blockDim.x + threadIdx.x;
    if (idx >= (size_t)NTOK * DM) return;
    int d = idx % DM;
    int r = idx / DM;
    int c = (r % SEQ) & 7;
    resid[idx] += ce[c * DM + d];
}

// LayerNorm over rows; optional bf16 and/or f32 outputs
__global__ __launch_bounds__(256)
void vm_layernorm(const float* __restrict__ in, const float* __restrict__ w,
                  const float* __restrict__ b, bf16_t* __restrict__ out_bf,
                  float* __restrict__ out_f, int cols)
{
    __shared__ float s1[256], s2[256];
    const int r = blockIdx.x;
    const float* x = in + (size_t)r * cols;
    float sum = 0.f, sq = 0.f;
    for (int c = threadIdx.x; c < cols; c += 256) { float v = x[c]; sum += v; sq += v * v; }
    s1[threadIdx.x] = sum; s2[threadIdx.x] = sq;
    __syncthreads();
    for (int s = 128; s > 0; s >>= 1) {
        if ((int)threadIdx.x < s) { s1[threadIdx.x] += s1[threadIdx.x + s]; s2[threadIdx.x] += s2[threadIdx.x + s]; }
        __syncthreads();
    }
    const float mu   = s1[0] / cols;
    const float var  = s2[0] / cols - mu * mu;
    const float rstd = rsqrtf(var + EPS_);
    for (int c = threadIdx.x; c < cols; c += 256) {
        float v = (x[c] - mu) * rstd * w[c] + b[c];
        if (out_bf) out_bf[(size_t)r * cols + c] = f2bf(v);
        if (out_f)  out_f [(size_t)r * cols + c] = v;
    }
}

// causal depthwise conv (k=4) + SiLU; emits f32 (for scan) + bf16 (for x_proj GEMM)
__global__ void vm_conv_silu(const float* __restrict__ xr, const float* __restrict__ cw,
                             const float* __restrict__ cb, float* __restrict__ xcf,
                             bf16_t* __restrict__ xcb)
{
    size_t idx = (size_t)blockIdx.x * blockDim.x + threadIdx.x;
    if (idx >= (size_t)NTOK * DI) return;
    int d = idx % DI;
    int r = idx / DI;
    int l = r % SEQ;
    float acc = cb[d];
#pragma unroll
    for (int k = 0; k < DCONV; ++k) {
        int ll = l - (DCONV - 1) + k;
        if (ll >= 0) acc += cw[d * DCONV + k] * xr[(size_t)(r - l + ll) * DI + d];
    }
    float s = siluf(acc);
    xcf[idx] = s;
    xcb[idx] = f2bf(s);
}

// dt = softplus(dbc[:, :48] @ dtw^T + dtb)
__global__ void vm_dtproj(const float* __restrict__ dbc, const float* __restrict__ dtw,
                          const float* __restrict__ dtb, float* __restrict__ dt)
{
    size_t idx = (size_t)blockIdx.x * blockDim.x + threadIdx.x;
    if (idx >= (size_t)NTOK * DI) return;
    int d = idx % DI;
    int r = idx / DI;
    const float* dr = dbc + (size_t)r * 80;
    const float* wd = dtw + (size_t)d * DTRANK;
    float acc = dtb[d];
#pragma unroll 8
    for (int j = 0; j < DTRANK; ++j) acc += dr[j] * wd[j];
    dt[idx] = (acc > 20.f) ? acc : log1pf(expf(acc));
}

// selective scan: one thread per (batch, channel); 16 states in registers
__global__ __launch_bounds__(256)
void vm_scan(const float* __restrict__ dt, const float* __restrict__ xcf,
             const float* __restrict__ dbc, const float* __restrict__ Alog,
             const float* __restrict__ Dp, float* __restrict__ y)
{
    const int d = blockIdx.x * blockDim.x + threadIdx.x;
    const int b = blockIdx.y;
    if (d >= DI) return;
    float A_[DSTATE], h[DSTATE];
#pragma unroll
    for (int n = 0; n < DSTATE; ++n) { A_[n] = -expf(Alog[(size_t)d * DSTATE + n]); h[n] = 0.f; }
    const float Dd = Dp[d];
    for (int l = 0; l < SEQ; ++l) {
        const size_t r = (size_t)b * SEQ + l;
        const float dtv = dt[r * DI + d];
        const float xv  = xcf[r * DI + d];
        const float dtx = dtv * xv;
        const float* Br = dbc + r * 80 + DTRANK;
        const float* Cr = dbc + r * 80 + DTRANK + DSTATE;
        float acc = 0.f;
#pragma unroll
        for (int n = 0; n < DSTATE; ++n) {
            h[n] = expf(dtv * A_[n]) * h[n] + dtx * Br[n];
            acc += h[n] * Cr[n];
        }
        y[r * DI + d] = acc + Dd * xv;
    }
}

// y' = LN(y)*w+b, out = y' * silu(z)  (bf16 for out_proj GEMM)
__global__ __launch_bounds__(256)
void vm_ssm_norm_gate(const float* __restrict__ y, const float* __restrict__ z,
                      const float* __restrict__ w, const float* __restrict__ b,
                      bf16_t* __restrict__ out)
{
    __shared__ float s1[256], s2[256];
    const int r = blockIdx.x;
    const float* yr = y + (size_t)r * DI;
    const float* zr = z + (size_t)r * DI;
    float sum = 0.f, sq = 0.f;
    for (int c = threadIdx.x; c < DI; c += 256) { float v = yr[c]; sum += v; sq += v * v; }
    s1[threadIdx.x] = sum; s2[threadIdx.x] = sq;
    __syncthreads();
    for (int s = 128; s > 0; s >>= 1) {
        if ((int)threadIdx.x < s) { s1[threadIdx.x] += s1[threadIdx.x + s]; s2[threadIdx.x] += s2[threadIdx.x + s]; }
        __syncthreads();
    }
    const float mu   = s1[0] / DI;
    const float var  = s2[0] / DI - mu * mu;
    const float rstd = rsqrtf(var + EPS_);
    for (int c = threadIdx.x; c < DI; c += 256) {
        float v = (yr[c] - mu) * rstd * w[c] + b[c];
        out[(size_t)r * DI + c] = f2bf(v * siluf(zr[c]));
    }
}

__global__ void vm_pool_mean(const float* __restrict__ hf, float* __restrict__ pooled)
{
    int idx = blockIdx.x * blockDim.x + threadIdx.x;
    if (idx >= BATCH_ * DM) return;
    int d = idx % DM, b = idx / DM;
    float s = 0.f;
    for (int l = 0; l < SEQ; ++l) s += hf[((size_t)b * SEQ + l) * DM + d];
    pooled[idx] = s * (1.0f / SEQ);
}

__global__ void vm_head(const float* __restrict__ pooled, const float* __restrict__ hw_,
                        const float* __restrict__ hb, float* __restrict__ out)
{
    int idx = blockIdx.x * blockDim.x + threadIdx.x;
    if (idx >= BATCH_ * NCLS) return;
    int c = idx % NCLS, b = idx / NCLS;
    const float* p = pooled + b * DM;
    const float* w = hw_ + (size_t)c * DM;
    float s = hb[c];
    for (int d = 0; d < DM; ++d) s += p[d] * w[d];
    out[idx] = s;
}

// ---------------------------------------------------------------------------
extern "C" void kernel_launch(void* const* d_in, const int* in_sizes, int n_in,
                              void* d_out, int out_size, void* d_ws, size_t ws_size,
                              hipStream_t stream)
{
    const float* x         = (const float*)d_in[0];
    const float* patch_w   = (const float*)d_in[1];
    const float* patch_b   = (const float*)d_in[2];
    const float* ce        = (const float*)d_in[3];
    const float* norm_w    = (const float*)d_in[4];
    const float* norm_b    = (const float*)d_in[5];
    const float* in_proj_w = (const float*)d_in[6];
    const float* conv_w    = (const float*)d_in[7];
    const float* conv_b    = (const float*)d_in[8];
    const float* x_proj_w  = (const float*)d_in[9];
    const float* dt_proj_w = (const float*)d_in[10];
    const float* dt_proj_b = (const float*)d_in[11];
    const float* A_log     = (const float*)d_in[12];
    const float* D_param   = (const float*)d_in[13];
    const float* snw       = (const float*)d_in[14];
    const float* snb       = (const float*)d_in[15];
    const float* out_proj_w= (const float*)d_in[16];
    const float* norm_f_w  = (const float*)d_in[17];
    const float* norm_f_b  = (const float*)d_in[18];
    const float* head_w    = (const float*)d_in[19];
    const float* head_b    = (const float*)d_in[20];
    float* out = (float*)d_out;

    char* ws = (char*)d_ws;
    size_t off = 0;
    auto alloc = [&](size_t bytes) -> void* {
        void* p = ws + off;
        off = (off + bytes + 255) & ~(size_t)255;
        return p;
    };

    float*  resid    = (float*) alloc((size_t)NTOK * DM * 4);
    float*  hid      = (float*) alloc((size_t)NTOK * DM * 4);
    bf16_t* hn_bf    = (bf16_t*)alloc((size_t)NTOK * DM * 2);
    float*  xr       = (float*) alloc((size_t)NTOK * DI * 4);   // raw x, later y
    float*  zbuf     = (float*) alloc((size_t)NTOK * DI * 4);   // z, later final hf
    float*  xconv_f  = (float*) alloc((size_t)NTOK * DI * 4);
    bf16_t* xconv_bf = (bf16_t*)alloc((size_t)NTOK * DI * 2);
    float*  dbc      = (float*) alloc((size_t)NTOK * 80 * 4);
    float*  dtbuf    = (float*) alloc((size_t)NTOK * DI * 4);
    bf16_t* ygate_bf = (bf16_t*)alloc((size_t)NTOK * DI * 2);
    bf16_t* Apatch   = (bf16_t*)alloc((size_t)NTOK * 256 * 2);
    bf16_t* wpatch   = (bf16_t*)alloc((size_t)DM * 256 * 2);
    bf16_t* w_in     = (bf16_t*)alloc((size_t)LAYERS * 2 * DI * DM * 2);
    bf16_t* w_xp     = (bf16_t*)alloc((size_t)LAYERS * 80 * DI * 2);
    bf16_t* w_out    = (bf16_t*)alloc((size_t)LAYERS * DM * DI * 2);
    float*  pooled   = (float*) alloc((size_t)BATCH_ * DM * 4);
    (void)ws_size; (void)in_sizes; (void)n_in; (void)out_size;

    auto cvt = [&](const float* src, bf16_t* dst, size_t n) {
        vm_f32_to_bf16<<<(unsigned)((n + 255) / 256), 256, 0, stream>>>(src, dst, n);
    };
    auto gemm = [&](const bf16_t* A, const bf16_t* W, const float* bias,
                    float* C, int M, int N, int K) {
        const dim3 gy(1, (M + 127) / 128);
        const int n64 = (N / 64) * 64;
        if (n64 > 0) {
            dim3 grid(n64 / 64, gy.y);
            vm_gemm_bf16_wmma<4><<<grid, 256, 0, stream>>>(A, W, bias, C, M, N, K, 0);
        }
        const int rem = (N - n64) / 16;
        if (rem == 1)
            vm_gemm_bf16_wmma<1><<<dim3(1, gy.y), 256, 0, stream>>>(A, W, bias, C, M, N, K, n64);
        else if (rem == 2)
            vm_gemm_bf16_wmma<2><<<dim3(1, gy.y), 256, 0, stream>>>(A, W, bias, C, M, N, K, n64);
        else if (rem == 3)
            vm_gemm_bf16_wmma<3><<<dim3(1, gy.y), 256, 0, stream>>>(A, W, bias, C, M, N, K, n64);
    };

    // ---- weight conversions (every call; deterministic) ----
    cvt(patch_w,    wpatch, (size_t)DM * 256);
    cvt(in_proj_w,  w_in,   (size_t)LAYERS * 2 * DI * DM);
    cvt(x_proj_w,   w_xp,   (size_t)LAYERS * 80 * DI);
    cvt(out_proj_w, w_out,  (size_t)LAYERS * DM * DI);

    // ---- patch embedding ----
    vm_im2col<<<(unsigned)(((size_t)NTOK * 256 + 255) / 256), 256, 0, stream>>>(x, Apatch);
    gemm(Apatch, wpatch, patch_b, resid, NTOK, DM, 256);
    vm_add_channel_embed<<<(unsigned)(((size_t)NTOK * DM + 255) / 256), 256, 0, stream>>>(resid, ce);

    const unsigned gDI = (unsigned)(((size_t)NTOK * DI + 255) / 256);

    for (int l = 0; l < LAYERS; ++l) {
        if (l > 0)
            vm_add_inplace<<<(unsigned)(((size_t)NTOK * DM + 255) / 256), 256, 0, stream>>>(
                resid, hid, (size_t)NTOK * DM);
        vm_layernorm<<<NTOK, 256, 0, stream>>>(resid, norm_w + l * DM, norm_b + l * DM,
                                               hn_bf, (float*)nullptr, DM);
        const bf16_t* Wl = w_in + (size_t)l * 2 * DI * DM;
        gemm(hn_bf, Wl,                   nullptr, xr,   NTOK, DI, DM);   // x half
        gemm(hn_bf, Wl + (size_t)DI * DM, nullptr, zbuf, NTOK, DI, DM);   // z half
        vm_conv_silu<<<gDI, 256, 0, stream>>>(xr, conv_w + (size_t)l * DI * DCONV,
                                              conv_b + l * DI, xconv_f, xconv_bf);
        gemm(xconv_bf, w_xp + (size_t)l * 80 * DI, nullptr, dbc, NTOK, 80, DI);
        vm_dtproj<<<gDI, 256, 0, stream>>>(dbc, dt_proj_w + (size_t)l * DI * DTRANK,
                                           dt_proj_b + l * DI, dtbuf);
        vm_scan<<<dim3((DI + 255) / 256, BATCH_), 256, 0, stream>>>(
            dtbuf, xconv_f, dbc, A_log + (size_t)l * DI * DSTATE, D_param + l * DI,
            xr /* y overlays raw-x */);
        vm_ssm_norm_gate<<<NTOK, 256, 0, stream>>>(xr, zbuf, snw + l * DI, snb + l * DI, ygate_bf);
        gemm(ygate_bf, w_out + (size_t)l * DM * DI, nullptr, hid, NTOK, DM, DI);
    }

    // ---- final LN + pool + head ----
    vm_add_inplace<<<(unsigned)(((size_t)NTOK * DM + 255) / 256), 256, 0, stream>>>(
        resid, hid, (size_t)NTOK * DM);
    vm_layernorm<<<NTOK, 256, 0, stream>>>(resid, norm_f_w, norm_f_b,
                                           (bf16_t*)nullptr, zbuf /* hf overlay */, DM);
    vm_pool_mean<<<(BATCH_ * DM + 255) / 256, 256, 0, stream>>>(zbuf, pooled);
    vm_head<<<(BATCH_ * NCLS + 255) / 256, 256, 0, stream>>>(pooled, head_w, head_b, out);
}
